// Raindrop_v2_22187801051700
// MI455X (gfx1250) — compile-verified, matrix operand order checked by reading, not run on hardware
//
#include <hip/hip_runtime.h>
#include <hip/hip_bf16.h>

typedef __attribute__((ext_vector_type(16))) _Float16 v16h;
typedef __attribute__((ext_vector_type(8)))  _Float16 v8h;
typedef __attribute__((ext_vector_type(8)))  float    v8f;

#define LSEQ    256
#define BATCH   256
#define NSENS   64
#define D_PE    16
#define D_MODEL 256
#define FDIM    1024
#define D_TR    272
#define D_TRP   288   /* 272 padded to multiple of 32 */
#define NHEAD   4
#define HDIM    68
#define HDP     96    /* 68 padded to multiple of 32 */
#define QKVD    816
#define QKVP    832   /* 816 padded to multiple of 32 */
#define D_FIN   336
#define TOK     (LSEQ*BATCH)

// ---------------------------------------------------------------------------
// WMMA fragment load: 16 rows x 32 K of f16 for one operand tile.
// Layout per CDNA5 ISA 7.12.2 (16-bit A 16x32): lane&15 = row,
// lane>>4 selects K-octets {0..7,16..23} vs {8..15,24..31}.
// All operands are padded so loads are unconditional, 16B-aligned b128s.
// ---------------------------------------------------------------------------
__device__ __forceinline__ v16h frag_ld(const _Float16* __restrict__ p, int row0,
                                        int k0, int ld) {
  int lane = threadIdx.x & 31;
  const _Float16* base =
      p + (size_t)(row0 + (lane & 15)) * ld + k0 + ((lane >> 4) & 1) * 8;
  v8h lo = *(const v8h*)(base);
  v8h hi = *(const v8h*)(base + 16);
  v16h f;
#pragma unroll
  for (int j = 0; j < 8; ++j) { f[j] = lo[j]; f[j + 8] = hi[j]; }
  return f;
}

// ---------------------------------------------------------------------------
// Batched WMMA GEMM:  C = A(MxK f16, row-major) * B + bias
// B given as Bt [N,K] row-major (weights/activations pre-transposed+padded).
// Each wave computes a 32x32 C block (2x2 WMMA tiles).
// Requirements (guaranteed by construction): M%32==0, N%32==0, K%32==0,
// zero padding in K tails and N tails. Store guarded by Nout.
// batch = blockIdx.y -> (outer,inner) two-level strides for (b,head) slices.
// ---------------------------------------------------------------------------
__global__ void gemm_wmma(const _Float16* __restrict__ A, const _Float16* __restrict__ Bm,
                          float* __restrict__ C, _Float16* __restrict__ C16,
                          const float* __restrict__ bias,
                          long aO, long aI, long bO, long bI, long cO, long cI,
                          long biasO, long biasI, int innerCnt,
                          int M, int N, int Nout, int K,
                          int lda, int ldb, int ldc, int relu) {
  int batch = blockIdx.y;
  int outer = batch / innerCnt;
  int inner = batch - outer * innerCnt;
  A  += outer * aO + inner * aI;
  Bm += outer * bO + inner * bI;
  long coff = outer * cO + inner * cI;
  if (C)    C   += coff;
  if (C16)  C16 += coff;
  if (bias) bias += outer * biasO + inner * biasI;

  int tilesN = N >> 5;
  int tilesM = M >> 5;
  int tile = blockIdx.x * (blockDim.x >> 5) + (threadIdx.x >> 5);
  if (tile >= tilesM * tilesN) return;  // wave-uniform
  int tm = tile / tilesN, tn = tile - tm * tilesN;
  int row0 = tm << 5, col0 = tn << 5;

  v8f acc00 = {}, acc01 = {}, acc10 = {}, acc11 = {};
  for (int k = 0; k < K; k += 32) {
    if (k + 32 < K) __builtin_prefetch(A + (size_t)row0 * lda + k + 32, 0, 1);
    v16h a0 = frag_ld(A, row0,      k, lda);
    v16h a1 = frag_ld(A, row0 + 16, k, lda);
    v16h b0 = frag_ld(Bm, col0,      k, ldb);
    v16h b1 = frag_ld(Bm, col0 + 16, k, ldb);
    acc00 = __builtin_amdgcn_wmma_f32_16x16x32_f16(false, a0, false, b0, (short)0, acc00, false, false);
    acc01 = __builtin_amdgcn_wmma_f32_16x16x32_f16(false, a0, false, b1, (short)0, acc01, false, false);
    acc10 = __builtin_amdgcn_wmma_f32_16x16x32_f16(false, a1, false, b0, (short)0, acc10, false, false);
    acc11 = __builtin_amdgcn_wmma_f32_16x16x32_f16(false, a1, false, b1, (short)0, acc11, false, false);
  }
  // C layout: VGPR r -> row = r + (lane>=16 ? 8:0), col = lane&15
  int lane = threadIdx.x & 31;
  int ln = lane & 15;
  int rsel = (lane & 16) ? 8 : 0;
#pragma unroll
  for (int ct = 0; ct < 2; ++ct) {
    int cn = col0 + ct * 16 + ln;
    if (cn >= Nout) continue;
    float bv = bias ? bias[cn] : 0.0f;
#pragma unroll
    for (int rt = 0; rt < 2; ++rt) {
      v8f acc = (rt == 0) ? (ct == 0 ? acc00 : acc01) : (ct == 0 ? acc10 : acc11);
      int rbase = row0 + rt * 16 + rsel;
#pragma unroll
      for (int r = 0; r < 8; ++r) {
        float v = acc[r] + bv;
        if (relu) v = fmaxf(v, 0.0f);
        size_t idx = (size_t)(rbase + r) * ldc + cn;
        if (C)   C[idx]   = v;
        if (C16) C16[idx] = (_Float16)v;
      }
    }
  }
}

// ---------------------------------------------------------------------------
// Elementwise / reduction kernels
// ---------------------------------------------------------------------------
__global__ void k_pe(const float* __restrict__ times, float* __restrict__ pe) {
  int idx = blockIdx.x * blockDim.x + threadIdx.x;
  if (idx >= LSEQ * BATCH) return;
  float t = times[idx];
  float* o = pe + (size_t)idx * D_PE;
#pragma unroll
  for (int i = 0; i < 8; ++i) {
    float ts = __powf(256.0f, (float)i * (1.0f / 7.0f));
    float s = t / ts;
    o[i]     = __sinf(s);
    o[i + 8] = __cosf(s);
  }
}

// xb[b,n,l*4+j] = relu(src[l,b,n] * R_u[n*4+j])  (f16)
__global__ void k_xb(const float* __restrict__ src, const float* __restrict__ Ru,
                     _Float16* __restrict__ xb) {
  int idx = blockIdx.x * blockDim.x + threadIdx.x;
  if (idx >= BATCH * NSENS * FDIM) return;
  int j = idx & 3;
  int l = (idx >> 2) & 255;
  int n = (idx >> 10) & 63;
  int b = idx >> 16;
  float v = src[((size_t)l * BATCH + b) * (2 * NSENS) + n];
  float h = v * Ru[n * 4 + j];
  xb[idx] = (_Float16)(h > 0.0f ? h : 0.0f);
}

__global__ void k_pmean(const float* __restrict__ pe, float* __restrict__ pm) {
  int b = blockIdx.x, d = threadIdx.x;
  if (d >= D_PE) return;
  float s = 0.f;
  for (int l = 0; l < LSEQ; ++l) s += pe[((size_t)l * BATCH + b) * D_PE + d];
  pm[b * D_PE + d] = s * (1.0f / LSEQ);
}

__global__ void k_emb(const float* __restrict__ st, const float* __restrict__ w,
                      const float* __restrict__ bb, float* __restrict__ emb) {
  int b = blockIdx.x, n = threadIdx.x;
  if (n >= 64) return;
  float s = bb[n];
#pragma unroll
  for (int d = 0; d < 9; ++d) s += st[b * 9 + d] * w[d * 64 + n];
  emb[b * 64 + n] = s;
}

// Pw[b,n] = pmean[b,:] . wp[:,n]
__global__ void k_pw(const float* __restrict__ pm, const float* __restrict__ wp,
                     float* __restrict__ Pw) {
  int idx = blockIdx.x * blockDim.x + threadIdx.x;
  if (idx >= BATCH * FDIM) return;
  int n = idx & 1023, b = idx >> 10;
  float s = 0.f;
#pragma unroll
  for (int d = 0; d < 16; ++d) s += pm[b * 16 + d] * wp[d * FDIM + n];
  Pw[idx] = s;
}

// out[n*Kpad+k] = (k<K && n<N) ? f16(in[k*N+n]) : 0   (weight transpose+pad)
__global__ void k_transpose(const float* __restrict__ in, _Float16* __restrict__ out,
                            int K, int N, int Kpad) {
  int n = blockIdx.x * 16 + threadIdx.x;
  int k = blockIdx.y * 16 + threadIdx.y;
  float v = (k < K && n < N) ? in[(size_t)k * N + n] : 0.0f;
  out[(size_t)n * Kpad + k] = (_Float16)v;
}

// batched: zT[b][n*64+k] = f16(zf[b][k*1024+n]),  n in [0,1024), k in [0,64)
__global__ void k_ztrans(const float* __restrict__ zf, _Float16* __restrict__ zT) {
  int n = blockIdx.x * 16 + threadIdx.x;
  int k = blockIdx.y * 16 + threadIdx.y;
  int b = blockIdx.z;
  zT[((size_t)b * FDIM + n) * NSENS + k] =
      (_Float16)zf[((size_t)b * NSENS + k) * FDIM + n];
}

// Pack Q,K into (B,H,L,96) zero-padded; V transposed into (B,H,96,L).
__global__ void k_qkv_pack(const _Float16* __restrict__ qkv, _Float16* __restrict__ Qp,
                           _Float16* __restrict__ Kp, _Float16* __restrict__ Vt) {
  int idx = blockIdx.x * blockDim.x + threadIdx.x;
  if (idx >= BATCH * NHEAD * LSEQ * HDP) return;
  int c  = idx % HDP;
  int l  = (idx / HDP) & 255;
  int bh = idx / (HDP * LSEQ);
  int h  = bh & 3;
  int b  = bh >> 2;
  _Float16 q = (_Float16)0.0f, k = (_Float16)0.0f, v = (_Float16)0.0f;
  if (c < HDIM) {
    size_t base = ((size_t)l * BATCH + b) * QKVD + h * HDIM + c;
    q = qkv[base];
    k = qkv[base + D_TR];
    v = qkv[base + 2 * D_TR];
  }
  Qp[idx] = q;
  Kp[idx] = k;
  Vt[(((size_t)bh * HDP) + c) * LSEQ + l] = v;
}

__global__ void k_fill0_h(_Float16* __restrict__ p, size_t count) {
  size_t i = (size_t)blockIdx.x * blockDim.x + threadIdx.x;
  if (i < count) p[i] = (_Float16)0.0f;
}

__global__ void k_zlzr(const float* __restrict__ z, const float* __restrict__ al,
                       const float* __restrict__ ar, float* __restrict__ zl,
                       float* __restrict__ zr) {
  __shared__ float s1[256], s2[256];
  int i = blockIdx.x;  // b*64 + node
  int t = threadIdx.x;
  const float* row = z + (size_t)i * FDIM;
  float a = 0.f, c = 0.f;
  for (int k = t; k < FDIM; k += 256) { float v = row[k]; a += v * al[k]; c += v * ar[k]; }
  s1[t] = a; s2[t] = c; __syncthreads();
  for (int off = 128; off > 0; off >>= 1) {
    if (t < off) { s1[t] += s1[t + off]; s2[t] += s2[t + off]; }
    __syncthreads();
  }
  if (t == 0) { zl[i] = s1[0]; zr[i] = s2[0]; }
}

// segment softmax over src per (b,dst); writes alpha (edge f32), edge f16, transposed f16
__global__ void k_score(const float* __restrict__ zl, const float* __restrict__ zr,
                        const float* __restrict__ ew, float* __restrict__ aE,
                        _Float16* __restrict__ aEh, _Float16* __restrict__ aT) {
  __shared__ float sh[64];
  int id = blockIdx.x;  // b*64 + d
  int b = id >> 6, d = id & 63;
  int s = threadIdx.x;  // 64 threads
  float sc = zl[b * 64 + s] + zr[id];
  sc = sc > 0.f ? sc : 0.2f * sc;
  float w = ew ? ew[((size_t)b << 12) + (s << 6) + d] : 1.0f;
  sc *= w;
  sh[s] = sc; __syncthreads();
  for (int off = 32; off > 0; off >>= 1) {
    if (s < off) sh[s] = fmaxf(sh[s], sh[s + off]);
    __syncthreads();
  }
  float m = sh[0]; __syncthreads();
  float e = __expf(sc - m);
  sh[s] = e; __syncthreads();
  for (int off = 32; off > 0; off >>= 1) {
    if (s < off) sh[s] += sh[s + off];
    __syncthreads();
  }
  float a = e / sh[0];
  size_t eidx = ((size_t)b << 12) + (s << 6) + d;
  aE[eidx]  = a;
  aEh[eidx] = (_Float16)a;
  aT[((size_t)b << 12) + (d << 6) + s] = (_Float16)a;
}

// transformer input: X (stride 272 f32) + Xh (stride 288 f16, zero pad)
__global__ void k_build_x0(const float* __restrict__ o2, const float* __restrict__ pe,
                           float* __restrict__ X, _Float16* __restrict__ Xh) {
  size_t idx = (size_t)blockIdx.x * blockDim.x + threadIdx.x;
  if (idx >= (size_t)TOK * D_TRP) return;
  int c = (int)(idx % D_TRP);
  size_t t = idx / D_TRP;
  if (c >= D_TR) { Xh[idx] = (_Float16)0.0f; return; }
  int b = (int)(t & 255);
  int l = (int)(t >> 8);
  float v;
  if (c < D_MODEL) {
    int n = c >> 2, j = c & 3;
    v = o2[(((size_t)b * NSENS + n) << 10) + (l << 2) + j];
  } else {
    v = pe[((size_t)l * BATCH + b) * D_PE + (c - D_MODEL)];
  }
  X[t * D_TR + c] = v;
  Xh[idx] = (_Float16)v;
}

// masked scaled softmax over key dim; f16 probabilities out
__global__ void k_attn_softmax(const float* __restrict__ S, _Float16* __restrict__ P,
                               const int* __restrict__ lengths) {
  __shared__ float sh[256];
  int r = blockIdx.x;       // (b*H+h)*L + q
  int bh = r >> 8;
  int b = bh >> 2;
  int k = threadIdx.x;
  int len = lengths[b];
  const float scale = 0.121267813f;  // 1/sqrt(68)
  float v = S[(size_t)r * LSEQ + k] * scale;
  if (k >= len) v = -1.0e9f;
  sh[k] = v; __syncthreads();
  for (int off = 128; off > 0; off >>= 1) {
    if (k < off) sh[k] = fmaxf(sh[k], sh[k + off]);
    __syncthreads();
  }
  float m = sh[0]; __syncthreads();
  float e = __expf(v - m);
  sh[k] = e; __syncthreads();
  for (int off = 128; off > 0; off >>= 1) {
    if (k < off) sh[k] += sh[k + off];
    __syncthreads();
  }
  P[(size_t)r * LSEQ + k] = (_Float16)(e / sh[0]);
}

// X = LN(res + dlt) * g + b ; writes X (272) + Xh (288, zero pad)
__global__ void k_add_ln(const float* __restrict__ res, const float* __restrict__ dlt,
                         const float* __restrict__ g, const float* __restrict__ bb,
                         float* __restrict__ X, _Float16* __restrict__ Xh) {
  __shared__ float s1[256], s2[256];
  size_t t = blockIdx.x;
  int tid = threadIdx.x;
  const float* r  = res + t * D_TR;
  const float* dd = dlt + t * D_TR;
  float vals[2];
  float a = 0.f, q = 0.f;
  for (int c = tid, it = 0; c < D_TR; c += 256, ++it) {
    float v = r[c] + dd[c];
    vals[it] = v; a += v; q += v * v;
  }
  s1[tid] = a; s2[tid] = q; __syncthreads();
  for (int off = 128; off > 0; off >>= 1) {
    if (tid < off) { s1[tid] += s1[tid + off]; s2[tid] += s2[tid + off]; }
    __syncthreads();
  }
  float mean = s1[0] * (1.0f / D_TR);
  float var  = s2[0] * (1.0f / D_TR) - mean * mean;
  float inv  = rsqrtf(var + 1e-5f);
  for (int c = tid, it = 0; c < D_TR; c += 256, ++it) {
    float v = (vals[it] - mean) * inv * g[c] + bb[c];
    X[t * D_TR + c] = v;
    Xh[t * D_TRP + c] = (_Float16)v;
  }
  if (tid < D_TRP - D_TR) Xh[t * D_TRP + D_TR + tid] = (_Float16)0.0f;
}

__global__ void k_agg(const float* __restrict__ X, const int* __restrict__ lengths,
                      const float* __restrict__ emb, float* __restrict__ agg) {
  int b = blockIdx.x;
  int c = threadIdx.x;
  int len = lengths[b];
  if (c < D_TR) {
    float s = 0.f;
    for (int l = 0; l < len; ++l) s += X[((size_t)l * BATCH + b) * D_TR + c];
    agg[b * D_FIN + c] = s / ((float)len + 1.0f);
  } else if (c < D_FIN) {
    agg[b * D_FIN + c] = emb[b * 64 + (c - D_TR)];
  }
}

__global__ void k_mlp_hidden(const float* __restrict__ agg, const float* __restrict__ w1,
                             const float* __restrict__ b1, float* __restrict__ hid) {
  __shared__ float sa[D_FIN];
  int b = blockIdx.x, j = threadIdx.x;
  for (int c = j; c < D_FIN; c += blockDim.x) sa[c] = agg[b * D_FIN + c];
  __syncthreads();
  if (j < D_FIN) {
    float s = b1[j];
    for (int i = 0; i < D_FIN; ++i) s += sa[i] * w1[i * D_FIN + j];
    hid[b * D_FIN + j] = fmaxf(s, 0.f);
  }
}

__global__ void k_logits(const float* __restrict__ hid, const float* __restrict__ w2,
                         const float* __restrict__ b2, float* __restrict__ out) {
  int b = blockIdx.x, j = threadIdx.x;
  if (j < 2) {
    float s = b2[j];
    for (int i = 0; i < D_FIN; ++i) s += hid[b * D_FIN + i] * w2[i * 2 + j];
    out[b * 2 + j] = s;
  }
}

__global__ void k_zero1(float* p) {
  if (threadIdx.x == 0 && blockIdx.x == 0) p[0] = 0.f;
}

// distance = mean_{i,j} sqrt(max(G_ii + G_jj - 2 G_ij, 0))
__global__ void k_distance(const float* __restrict__ G, float* __restrict__ dist) {
  __shared__ float sh[256];
  int i = blockIdx.x, j = threadIdx.x;
  float gii = G[i * 257];
  float gjj = G[j * 257];
  float gij = G[i * 256 + j];
  float d2 = gii + gjj - 2.0f * gij;
  sh[j] = sqrtf(fmaxf(d2, 0.0f));
  __syncthreads();
  for (int off = 128; off > 0; off >>= 1) {
    if (j < off) sh[j] += sh[j + off];
    __syncthreads();
  }
  if (j == 0) atomicAdd(dist, sh[0] * (1.0f / 65536.0f));
}

// ---------------------------------------------------------------------------
// Host side
// ---------------------------------------------------------------------------
static inline void gemmF(hipStream_t s, int batches, int inner,
                         const void* A, long aO, long aI, int lda,
                         const void* B, long bO, long bI, int ldb,
                         void* C, void* C16, long cO, long cI, int ldc,
                         const void* bias, long biasO, long biasI,
                         int M, int N, int Nout, int K, int relu) {
  int tiles = (M >> 5) * (N >> 5);
  dim3 g((unsigned)((tiles + 7) / 8), (unsigned)batches, 1);
  dim3 blk(256, 1, 1);
  gemm_wmma<<<g, blk, 0, s>>>((const _Float16*)A, (const _Float16*)B, (float*)C,
      (_Float16*)C16, (const float*)bias, aO, aI, bO, bI, cO, cI, biasO, biasI,
      inner, M, N, Nout, K, lda, ldb, ldc, relu);
}

extern "C" void kernel_launch(void* const* d_in, const int* in_sizes, int n_in,
                              void* d_out, int out_size, void* d_ws, size_t ws_size,
                              hipStream_t stream) {
  (void)in_sizes; (void)n_in; (void)out_size; (void)ws_size;
  const float* src     = (const float*)d_in[0];
  const float* stat    = (const float*)d_in[1];
  const float* times   = (const float*)d_in[2];
  const int*   lengths = (const int*)  d_in[3];
  const float* R_u     = (const float*)d_in[4];
  const float* emb_w   = (const float*)d_in[5];
  const float* emb_b   = (const float*)d_in[6];
  const float* op1_W   = (const float*)d_in[7];
  const float* op1_al  = (const float*)d_in[8];
  const float* op1_ar  = (const float*)d_in[9];
  const float* op1_wp  = (const float*)d_in[10];
  const float* op2_W   = (const float*)d_in[11];
  const float* op2_al  = (const float*)d_in[12];
  const float* op2_ar  = (const float*)d_in[13];
  const float* op2_wp  = (const float*)d_in[14];
  const float* tr_wqkv = (const float*)d_in[15];
  const float* tr_bqkv = (const float*)d_in[16];
  const float* tr_wo   = (const float*)d_in[17];
  const float* tr_bo   = (const float*)d_in[18];
  const float* tr_ln1g = (const float*)d_in[19];
  const float* tr_ln1b = (const float*)d_in[20];
  const float* tr_w1   = (const float*)d_in[21];
  const float* tr_b1   = (const float*)d_in[22];
  const float* tr_w2   = (const float*)d_in[23];
  const float* tr_b2   = (const float*)d_in[24];
  const float* tr_ln2g = (const float*)d_in[25];
  const float* tr_ln2b = (const float*)d_in[26];
  const float* mlp_w1  = (const float*)d_in[27];
  const float* mlp_b1  = (const float*)d_in[28];
  const float* mlp_w2  = (const float*)d_in[29];
  const float* mlp_b2  = (const float*)d_in[30];
  float* out = (float*)d_out;  // [0..511] logits, [512] distance

  char* ws = (char*)d_ws;
  size_t off = 0;
  auto alloc = [&](size_t bytes) -> void* {
    void* p = ws + off;
    off += (bytes + 255) & ~(size_t)255;
    return p;
  };

  float*    pe    = (float*)   alloc((size_t)TOK * D_PE * 4);
  float*    pm    = (float*)   alloc((size_t)BATCH * D_PE * 4);
  float*    emb   = (float*)   alloc((size_t)BATCH * 64 * 4);
  _Float16* xb    = (_Float16*)alloc((size_t)BATCH * NSENS * FDIM * 2);
  _Float16* Wt1   = (_Float16*)alloc((size_t)FDIM * FDIM * 2);
  _Float16* Wt2   = (_Float16*)alloc((size_t)FDIM * FDIM * 2);
  _Float16* WqkvT = (_Float16*)alloc((size_t)2 * QKVP * D_TRP * 2);
  _Float16* WoT   = (_Float16*)alloc((size_t)2 * D_TRP * D_TRP * 2);
  _Float16* W1t   = (_Float16*)alloc((size_t)2 * 1024 * D_TRP * 2);
  _Float16* W2t   = (_Float16*)alloc((size_t)2 * D_TRP * 1024 * 2);
  float*    Pw1   = (float*)   alloc((size_t)BATCH * FDIM * 4);
  float*    Pw2   = (float*)   alloc((size_t)BATCH * FDIM * 4);
  float*    zl    = (float*)   alloc((size_t)BATCH * 64 * 4);
  float*    zr    = (float*)   alloc((size_t)BATCH * 64 * 4);
  float*    aE1   = (float*)   alloc((size_t)BATCH * 4096 * 4);
  float*    aE2   = (float*)   alloc((size_t)BATCH * 4096 * 4);
  _Float16* aEh   = (_Float16*)alloc((size_t)BATCH * 4096 * 2);
  _Float16* aT    = (_Float16*)alloc((size_t)BATCH * 4096 * 2);
  float*    G     = (float*)   alloc((size_t)256 * 256 * 4);
  float*    zf    = (float*)   alloc((size_t)BATCH * NSENS * FDIM * 4);
  _Float16* zT    = (_Float16*)alloc((size_t)BATCH * FDIM * NSENS * 2);
  _Float16* o1h   = (_Float16*)alloc((size_t)BATCH * NSENS * FDIM * 2);
  float*    o2f   = (float*)   alloc((size_t)BATCH * NSENS * FDIM * 4);
  float*    X     = (float*)   alloc((size_t)TOK * D_TR * 4);
  _Float16* Xh    = (_Float16*)alloc((size_t)TOK * D_TRP * 2);
  _Float16* QKVh  = (_Float16*)alloc((size_t)TOK * QKVD * 2);
  _Float16* Qp    = (_Float16*)alloc((size_t)BATCH * NHEAD * LSEQ * HDP * 2);
  _Float16* Kp    = (_Float16*)alloc((size_t)BATCH * NHEAD * LSEQ * HDP * 2);
  _Float16* Vt    = (_Float16*)alloc((size_t)BATCH * NHEAD * HDP * LSEQ * 2);
  float*    Sf    = (float*)   alloc((size_t)BATCH * NHEAD * LSEQ * LSEQ * 4);
  _Float16* Ph    = (_Float16*)alloc((size_t)BATCH * NHEAD * LSEQ * LSEQ * 2);
  _Float16* ATTh  = (_Float16*)alloc((size_t)TOK * D_TRP * 2);
  float*    PROJ  = (float*)   alloc((size_t)TOK * D_TR * 4);
  _Float16* H1h   = (_Float16*)alloc((size_t)TOK * 1024 * 2);
  float*    agg   = (float*)   alloc((size_t)BATCH * D_FIN * 4);
  float*    hid   = (float*)   alloc((size_t)BATCH * D_FIN * 4);

  // ---- prep ----
  k_pe<<<(TOK + 255) / 256, 256, 0, stream>>>(times, pe);
  k_xb<<<(BATCH * NSENS * FDIM) / 256, 256, 0, stream>>>(src, R_u, xb);
  k_pmean<<<BATCH, 32, 0, stream>>>(pe, pm);
  k_emb<<<BATCH, 64, 0, stream>>>(stat, emb_w, emb_b, emb);

  dim3 tb(16, 16);
  k_transpose<<<dim3(64, 64), tb, 0, stream>>>(op1_W, Wt1, FDIM, FDIM, FDIM);
  k_transpose<<<dim3(64, 64), tb, 0, stream>>>(op2_W, Wt2, FDIM, FDIM, FDIM);
  for (int l = 0; l < 2; ++l) {
    k_transpose<<<dim3(QKVP / 16, D_TRP / 16), tb, 0, stream>>>(
        tr_wqkv + (size_t)l * D_TR * QKVD, WqkvT + (size_t)l * QKVP * D_TRP,
        D_TR, QKVD, D_TRP);
    k_transpose<<<dim3(D_TRP / 16, D_TRP / 16), tb, 0, stream>>>(
        tr_wo + (size_t)l * D_TR * D_TR, WoT + (size_t)l * D_TRP * D_TRP,
        D_TR, D_TR, D_TRP);
    k_transpose<<<dim3(64, D_TRP / 16), tb, 0, stream>>>(
        tr_w1 + (size_t)l * D_TR * 1024, W1t + (size_t)l * 1024 * D_TRP,
        D_TR, 1024, D_TRP);
    k_transpose<<<dim3(D_TRP / 16, 64), tb, 0, stream>>>(
        tr_w2 + (size_t)l * 1024 * D_TR, W2t + (size_t)l * D_TRP * 1024,
        1024, D_TR, 1024);
  }
  k_pw<<<(BATCH * FDIM) / 256, 256, 0, stream>>>(pm, op1_wp, Pw1);
  k_pw<<<(BATCH * FDIM) / 256, 256, 0, stream>>>(pm, op2_wp, Pw2);
  k_fill0_h<<<(unsigned)(((size_t)TOK * D_TRP + 255) / 256), 256, 0, stream>>>(
      ATTh, (size_t)TOK * D_TRP);

  // ---- obs_prop layer 1 ----
  gemmF(stream, BATCH, 1,
        xb, (long)NSENS * FDIM, 0, FDIM,
        Wt1, 0, 0, FDIM,
        zf, nullptr, (long)NSENS * FDIM, 0, FDIM,
        Pw1, FDIM, 0,
        NSENS, FDIM, FDIM, FDIM, 0);
  k_ztrans<<<dim3(64, 4, BATCH), tb, 0, stream>>>(zf, zT);
  k_zlzr<<<BATCH * NSENS, 256, 0, stream>>>(zf, op1_al, op1_ar, zl, zr);
  k_score<<<BATCH * NSENS, 64, 0, stream>>>(zl, zr, (const float*)nullptr, aE1, aEh, aT);
  gemmF(stream, BATCH, 1,
        aT, 4096, 0, NSENS,
        zT, (long)FDIM * NSENS, 0, NSENS,
        nullptr, o1h, (long)NSENS * FDIM, 0, FDIM,
        nullptr, 0, 0,
        NSENS, FDIM, FDIM, NSENS, 0);

  // ---- obs_prop layer 2 ----
  gemmF(stream, BATCH, 1,
        o1h, (long)NSENS * FDIM, 0, FDIM,
        Wt2, 0, 0, FDIM,
        zf, nullptr, (long)NSENS * FDIM, 0, FDIM,
        Pw2, FDIM, 0,
        NSENS, FDIM, FDIM, FDIM, 0);
  k_ztrans<<<dim3(64, 4, BATCH), tb, 0, stream>>>(zf, zT);
  k_zlzr<<<BATCH * NSENS, 256, 0, stream>>>(zf, op2_al, op2_ar, zl, zr);
  k_score<<<BATCH * NSENS, 64, 0, stream>>>(zl, zr, aE1, aE2, aEh, aT);
  gemmF(stream, BATCH, 1,
        aT, 4096, 0, NSENS,
        zT, (long)FDIM * NSENS, 0, NSENS,
        o2f, nullptr, (long)NSENS * FDIM, 0, FDIM,
        nullptr, 0, 0,
        NSENS, FDIM, FDIM, NSENS, 0);

  // ---- distance: G = a a^T over (B, 4096) ----
  gemmF(stream, 1, 1,
        aEh, 0, 0, 4096,
        aEh, 0, 0, 4096,
        G, nullptr, 0, 0, 256,
        nullptr, 0, 0,
        256, 256, 256, 4096, 0);
  k_zero1<<<1, 32, 0, stream>>>(out + 512);
  k_distance<<<256, 256, 0, stream>>>(G, out + 512);

  // ---- transformer ----
  k_build_x0<<<(unsigned)(((size_t)TOK * D_TRP + 255) / 256), 256, 0, stream>>>(
      o2f, pe, X, Xh);

  for (int l = 0; l < 2; ++l) {
    // QKV = Xh @ WqkvT + bqkv  -> f16 only (stride 816)
    gemmF(stream, 1, 1,
          Xh, 0, 0, D_TRP,
          WqkvT + (size_t)l * QKVP * D_TRP, 0, 0, D_TRP,
          nullptr, QKVh, 0, 0, QKVD,
          tr_bqkv + (size_t)l * QKVD, 0, 0,
          TOK, QKVP, QKVD, D_TRP, 0);
    // pack heads: Qp,Kp (B,H,L,96), Vt (B,H,96,L)
    k_qkv_pack<<<(BATCH * NHEAD * LSEQ * HDP) / 256, 256, 0, stream>>>(QKVh, Qp, Kp, Vt);
    // S = Q K^T   (per (b,h))
    gemmF(stream, BATCH * NHEAD, NHEAD,
          Qp, (long)NHEAD * LSEQ * HDP, (long)LSEQ * HDP, HDP,
          Kp, (long)NHEAD * LSEQ * HDP, (long)LSEQ * HDP, HDP,
          Sf, nullptr, (long)NHEAD * LSEQ * LSEQ, (long)LSEQ * LSEQ, LSEQ,
          nullptr, 0, 0,
          LSEQ, LSEQ, LSEQ, HDP, 0);
    k_attn_softmax<<<BATCH * NHEAD * LSEQ, 256, 0, stream>>>(Sf, Ph, lengths);
    // O = P V^T'  (per (b,h)); compute 96 cols, store 68 into token-major ATTh
    gemmF(stream, BATCH * NHEAD, NHEAD,
          Ph, (long)NHEAD * LSEQ * LSEQ, (long)LSEQ * LSEQ, LSEQ,
          Vt, (long)NHEAD * HDP * LSEQ, (long)HDP * LSEQ, LSEQ,
          nullptr, ATTh, (long)D_TRP, (long)HDIM, BATCH * D_TRP,
          nullptr, 0, 0,
          LSEQ, HDP, HDIM, LSEQ, 0);
    // proj
    gemmF(stream, 1, 1,
          ATTh, 0, 0, D_TRP,
          WoT + (size_t)l * D_TRP * D_TRP, 0, 0, D_TRP,
          PROJ, nullptr, 0, 0, D_TR,
          tr_bo + (size_t)l * D_TR, 0, 0,
          TOK, D_TRP, D_TR, D_TRP, 0);
    k_add_ln<<<TOK, 256, 0, stream>>>(X, PROJ, tr_ln1g + (size_t)l * D_TR,
                                      tr_ln1b + (size_t)l * D_TR, X, Xh);
    // FFN
    gemmF(stream, 1, 1,
          Xh, 0, 0, D_TRP,
          W1t + (size_t)l * 1024 * D_TRP, 0, 0, D_TRP,
          nullptr, H1h, 0, 0, 1024,
          tr_b1 + (size_t)l * 1024, 0, 0,
          TOK, 1024, 1024, D_TRP, 1);
    gemmF(stream, 1, 1,
          H1h, 0, 0, 1024,
          W2t + (size_t)l * D_TRP * 1024, 0, 0, 1024,
          PROJ, nullptr, 0, 0, D_TR,
          tr_b2 + (size_t)l * D_TR, 0, 0,
          TOK, D_TRP, D_TR, 1024, 0);
    k_add_ln<<<TOK, 256, 0, stream>>>(X, PROJ, tr_ln2g + (size_t)l * D_TR,
                                      tr_ln2b + (size_t)l * D_TR, X, Xh);
  }

  // ---- pool + final MLP ----
  k_agg<<<BATCH, 352, 0, stream>>>(X, lengths, emb, agg);
  k_mlp_hidden<<<BATCH, 352, 0, stream>>>(agg, mlp_w1, mlp_b1, hid);
  k_logits<<<BATCH, 32, 0, stream>>>(hid, mlp_w2, mlp_b2, out);
}
